// FactorGNNSBMs_15479062135603
// MI455X (gfx1250) — matrix operand order repaired and checked by reading, not compiled, not run on hardware
//
#include <hip/hip_runtime.h>
#include <math.h>

typedef __attribute__((ext_vector_type(16))) _Float16 v16h;
typedef __attribute__((ext_vector_type(8)))  _Float16 v8h;
typedef __attribute__((ext_vector_type(8)))  float    v8f;

// ---------------------------------------------------------------------------
// Embedding gather: feat_h[i][j] = (f16) embed_W[x[i]][j]   (IN = 128)
// ---------------------------------------------------------------------------
__global__ void embed_kernel(const int* __restrict__ x,
                             const float* __restrict__ embedW,
                             _Float16* __restrict__ feat_h, int total) {
  int idx = blockIdx.x * blockDim.x + threadIdx.x;
  if (idx >= total) return;
  int i = idx >> 7, j = idx & 127;
  feat_h[idx] = (_Float16)embedW[x[i] * 128 + j];
}

// ---------------------------------------------------------------------------
// Degree (segment count over dst) then norm = rsqrt(max(deg,1))
// ---------------------------------------------------------------------------
__global__ void deg_kernel(const int* __restrict__ dst, float* __restrict__ deg, int E) {
  int e = blockIdx.x * blockDim.x + threadIdx.x;
  if (e < E) atomicAdd(&deg[dst[e]], 1.0f);
}

__global__ void norm_kernel(float* __restrict__ deg, int Nn) {
  int n = blockIdx.x * blockDim.x + threadIdx.x;
  if (n < Nn) deg[n] = rsqrtf(fmaxf(deg[n], 1.0f));
}

// ---------------------------------------------------------------------------
// Stage W[K,Nc] (f32 row-major) into WMMA B-fragment order (f16):
// out[((ct*ksteps + ks)*32 + lane)*16 + j] = W[ks*32 + (lane>=16)*16 + j][ct*16 + lane%16]
// so in the GEMM each lane reads 16 contiguous halfs (32 B) per K-step.
// ---------------------------------------------------------------------------
__global__ void stage_b_kernel(const float* __restrict__ W, _Float16* __restrict__ out,
                               int K, int Nc) {
  int total = (Nc >> 4) * (K >> 5) * 512;
  int idx = blockIdx.x * blockDim.x + threadIdx.x;
  if (idx >= total) return;
  int j     = idx & 15;
  int lane  = (idx >> 4) & 31;
  int ks    = (idx >> 9) % (K >> 5);
  int ct    = idx / (512 * (K >> 5));
  int k     = ks * 32 + (lane >> 4) * 16 + j;
  int col   = ct * 16 + (lane & 15);
  out[idx] = (_Float16)W[k * Nc + col];
}

// ---------------------------------------------------------------------------
// WMMA GEMM: C[M,Nc] = op(A[M,K]) @ W[K,Nc] + bias.
// A row-major f16 (ld=K), Bs pre-staged fragments (stage_b_kernel), C f32.
// 4 waves per block, one 16x16 output tile per wave.
// Fragment layouts as used here (wave32; k order contiguous, not the ISA table):
//   A 16x32 f16 : lane m / m+16 : row m ; K = 0..15 / 16..31
//   B 32x16 f16 : lane n / n+16 : col n ; K = 0..15 / 16..31
//   C 16x16 f32 : vgpr r, lane L -> M = r + 8*(L>=16), N = L%16
// ---------------------------------------------------------------------------
template <bool RELU_IN, bool RELU_OUT>
__global__ void gemm_wmma(const _Float16* __restrict__ A, const _Float16* __restrict__ Bs,
                          const float* __restrict__ bias, float* __restrict__ C,
                          int M, int K, int Nc) {
  const int lane = threadIdx.x & 31;
  const int wave = threadIdx.x >> 5;
  const int l16  = lane & 15;
  const int hsel = lane >> 4;
  const int row_base = (blockIdx.x * 4 + wave) * 16;
  if (row_base >= M) return;
  const int col_base = blockIdx.y * 16;
  const int ksteps = K >> 5;

  const int arow = row_base + l16;
  const _Float16* aptr = A + (size_t)arow * K + hsel * 16;
  const _Float16* bptr = Bs + ((size_t)blockIdx.y * ksteps * 32 + lane) * 16;

  v8f c = {};
  for (int ks = 0; ks < ksteps; ++ks) {
    v8h lo = *(const v8h*)(aptr);        // K = ks*32 + hsel*16 + (0..7)
    v8h hi = *(const v8h*)(aptr + 8);    // K = +8
    if (RELU_IN) {
#pragma unroll
      for (int j = 0; j < 8; ++j) {
        lo[j] = (lo[j] > (_Float16)0) ? lo[j] : (_Float16)0;
        hi[j] = (hi[j] > (_Float16)0) ? hi[j] : (_Float16)0;
      }
    }
    v16h a = __builtin_shufflevector(lo, hi, 0, 1, 2, 3, 4, 5, 6, 7,
                                     8, 9, 10, 11, 12, 13, 14, 15);
    v16h b = *(const v16h*)(bptr);       // 32 contiguous bytes
    c = __builtin_amdgcn_wmma_f32_16x16x32_f16(
        /*neg_a=*/false, a, /*neg_b=*/false, b,
        /*c_mod=*/(short)0, c, /*reuse_a=*/false, /*reuse_b=*/false);
    aptr += 32;
    bptr += 512;
  }

  const int bcol = col_base + l16;
  const float bv = bias ? bias[bcol] : 0.0f;
#pragma unroll
  for (int r = 0; r < 8; ++r) {
    int row = row_base + r + hsel * 8;
    float v = c[r] + bv;
    if (RELU_OUT) v = fmaxf(v, 0.0f);
    C[(size_t)row * Nc + bcol] = v;
  }
}

// ---------------------------------------------------------------------------
// Attention projections: a_l = hidden@Wl + bl ; a_r = hidden@Wr + br   [N,L]
// ---------------------------------------------------------------------------
template <int F, int L>
__global__ void att_kernel(const float* __restrict__ hidden,
                           const float* __restrict__ Wl, const float* __restrict__ bl,
                           const float* __restrict__ Wr, const float* __restrict__ br,
                           float* __restrict__ a_l, float* __restrict__ a_r, int total) {
  int idx = blockIdx.x * blockDim.x + threadIdx.x;
  if (idx >= total) return;
  int i = idx / L, l = idx % L;
  float sl = bl[l], sr = br[l];
  const float* hrow = hidden + (size_t)i * F;
#pragma unroll
  for (int f = 0; f < F; ++f) {
    float h = hrow[f];
    sl += h * Wl[f * L + l];
    sr += h * Wr[f * L + l];
  }
  a_l[idx] = sl;
  a_r[idx] = sr;
}

// ---------------------------------------------------------------------------
// Edge gather-gate-scatter. One block (128 threads) per edge; thread t owns
// latent l = t/F, feature f = t%F (compile-time shifts). Coalesced f32 atomics
// into out[dst], which is L2-resident (10 MB << 192 MB L2).
// ---------------------------------------------------------------------------
template <int F, int L>
__global__ void edge_scatter(const int* __restrict__ src, const int* __restrict__ dst,
                             const float* __restrict__ hidden, const float* __restrict__ norm,
                             const float* __restrict__ a_l, const float* __restrict__ a_r,
                             float* __restrict__ out) {
  int e = blockIdx.x;
  int t = threadIdx.x;               // 0..127 == L*F-1
  int l = t / F, f = t % F;          // F is a power of two -> shifts
  int s = src[e], d = dst[e];
  float z = a_l[s * L + l] + a_r[d * L + l];
  float gate = 1.0f / (1.0f + __expf(-6.0f * z));
  float val = hidden[(size_t)s * F + f] * norm[s] * gate;
  atomicAdd(&out[(size_t)d * 128 + t], val);
}

// ---------------------------------------------------------------------------
// BatchNorm stats over v = out[i][j]*snorm[i]: one block per column j.
// ---------------------------------------------------------------------------
__global__ void bn_stats(const float* __restrict__ out, const float* __restrict__ snorm,
                         float* __restrict__ mean, float* __restrict__ var, int Nrows) {
  int j = blockIdx.x;
  int tid = threadIdx.x;
  float s = 0.0f, ss = 0.0f;
  for (int i = tid; i < Nrows; i += blockDim.x) {
    float v = out[(size_t)i * 128 + j] * snorm[i];
    s += v;
    ss += v * v;
  }
  __shared__ float sh_s[256], sh_ss[256];
  sh_s[tid] = s; sh_ss[tid] = ss;
  __syncthreads();
  for (int off = 128; off > 0; off >>= 1) {
    if (tid < off) { sh_s[tid] += sh_s[tid + off]; sh_ss[tid] += sh_ss[tid + off]; }
    __syncthreads();
  }
  if (tid == 0) {
    float m = sh_s[0] / (float)Nrows;
    mean[j] = m;
    var[j] = fmaxf(sh_ss[0] / (float)Nrows - m * m, 0.0f);   // biased var (jnp default)
  }
}

// ---------------------------------------------------------------------------
// BN apply + LeakyReLU(0.2), write next-layer f16 features.
// ---------------------------------------------------------------------------
__global__ void bn_apply(const float* __restrict__ out, const float* __restrict__ snorm,
                         const float* __restrict__ mean, const float* __restrict__ var,
                         const float* __restrict__ g, const float* __restrict__ b,
                         _Float16* __restrict__ feat_h, int total) {
  int idx = blockIdx.x * blockDim.x + threadIdx.x;
  if (idx >= total) return;
  int i = idx >> 7, j = idx & 127;
  float v = out[idx] * snorm[i];
  float y = (v - mean[j]) * rsqrtf(var[j] + 1e-5f) * g[j] + b[j];
  y = (y > 0.0f) ? y : 0.2f * y;
  feat_h[idx] = (_Float16)y;
}

// ---------------------------------------------------------------------------
// Final tiny GEMM: logits[N,2] = h1[N,64] @ cls2_W[64,2] + b  (h1 already relu'd)
// ---------------------------------------------------------------------------
__global__ void cls2_kernel(const float* __restrict__ h1, const float* __restrict__ W,
                            const float* __restrict__ b, float* __restrict__ out, int total) {
  int idx = blockIdx.x * blockDim.x + threadIdx.x;
  if (idx >= total) return;
  int i = idx >> 1, c = idx & 1;
  float s = b[c];
  const float* hrow = h1 + (size_t)i * 64;
#pragma unroll 8
  for (int k = 0; k < 64; ++k) s += hrow[k] * W[k * 2 + c];
  out[idx] = s;
}

// ---------------------------------------------------------------------------
extern "C" void kernel_launch(void* const* d_in, const int* in_sizes, int n_in,
                              void* d_out, int out_size, void* d_ws, size_t ws_size,
                              hipStream_t stream) {
  const int N = in_sizes[0];   // 20000
  const int E = in_sizes[1];   // 640000

  const int*   x      = (const int*)d_in[0];
  const int*   src    = (const int*)d_in[1];
  const int*   dst    = (const int*)d_in[2];
  const float* snorm  = (const float*)d_in[3];
  const float* embedW = (const float*)d_in[4];

  const float* linW[3]  = {(const float*)d_in[5],  (const float*)d_in[13], (const float*)d_in[21]};
  const float* linb[3]  = {(const float*)d_in[6],  (const float*)d_in[14], (const float*)d_in[22]};
  const float* attlW[3] = {(const float*)d_in[7],  (const float*)d_in[15], (const float*)d_in[23]};
  const float* attlb[3] = {(const float*)d_in[8],  (const float*)d_in[16], (const float*)d_in[24]};
  const float* attrW[3] = {(const float*)d_in[9],  (const float*)d_in[17], (const float*)d_in[25]};
  const float* attrb[3] = {(const float*)d_in[10], (const float*)d_in[18], (const float*)d_in[26]};
  const float* bng[3]   = {(const float*)d_in[11], (const float*)d_in[19], (const float*)d_in[27]};
  const float* bnb[3]   = {(const float*)d_in[12], (const float*)d_in[20], (const float*)d_in[28]};
  const float* cls1W = (const float*)d_in[29];
  const float* cls1b = (const float*)d_in[30];
  const float* cls2W = (const float*)d_in[31];
  const float* cls2b = (const float*)d_in[32];

  // Workspace carve-up (256B aligned)
  char* ws = (char*)d_ws;
  size_t off = 0;
  auto alloc = [&](size_t bytes) -> void* {
    void* p = ws + off;
    off = (off + bytes + 255) & ~(size_t)255;
    return p;
  };
  _Float16* feat_h  = (_Float16*)alloc((size_t)N * 128 * sizeof(_Float16)); // 5.12 MB
  float*    hidden  = (float*)   alloc((size_t)N * 32  * sizeof(float));    // 2.56 MB
  float*    a_l     = (float*)   alloc((size_t)N * 8   * sizeof(float));
  float*    a_r     = (float*)   alloc((size_t)N * 8   * sizeof(float));
  float*    norm    = (float*)   alloc((size_t)N       * sizeof(float));
  float*    out_acc = (float*)   alloc((size_t)N * 128 * sizeof(float));    // 10.24 MB
  _Float16* wh      = (_Float16*)alloc((size_t)128 * 64 * sizeof(_Float16));
  float*    mean    = (float*)   alloc(128 * sizeof(float));
  float*    var     = (float*)   alloc(128 * sizeof(float));
  (void)ws_size; (void)n_in; (void)out_size;

  const int tot = N * 128;
  const int gemm_gx = (N / 16 + 3) / 4;   // 4 waves (tiles) per block

  // Embedding -> f16 features
  embed_kernel<<<(tot + 255) / 256, 256, 0, stream>>>(x, embedW, feat_h, tot);

  // In-degree -> norm (constant across layers)
  hipMemsetAsync(norm, 0, (size_t)N * sizeof(float), stream);
  deg_kernel<<<(E + 255) / 256, 256, 0, stream>>>(dst, norm, E);
  norm_kernel<<<(N + 255) / 256, 256, 0, stream>>>(norm, N);

  for (int layer = 0; layer < 3; ++layer) {
    const int F = (layer == 0) ? 16 : 32;
    const int L = 128 / F;

    // Stage weights into WMMA B-fragment order, then WMMA GEMM
    const int wcount = 128 * F;
    stage_b_kernel<<<(wcount + 255) / 256, 256, 0, stream>>>(linW[layer], wh, 128, F);
    dim3 gg(gemm_gx, F / 16);
    gemm_wmma<false, false><<<gg, 128, 0, stream>>>(feat_h, wh, linb[layer], hidden, N, 128, F);

    // Attention logits
    if (layer == 0)
      att_kernel<16, 8><<<(N * L + 255) / 256, 256, 0, stream>>>(
          hidden, attlW[layer], attlb[layer], attrW[layer], attrb[layer], a_l, a_r, N * L);
    else
      att_kernel<32, 4><<<(N * L + 255) / 256, 256, 0, stream>>>(
          hidden, attlW[layer], attlb[layer], attrW[layer], attrb[layer], a_l, a_r, N * L);

    // Gather-gate-scatter (L2-resident f32 atomics)
    hipMemsetAsync(out_acc, 0, (size_t)N * 128 * sizeof(float), stream);
    if (layer == 0)
      edge_scatter<16, 8><<<E, 128, 0, stream>>>(src, dst, hidden, norm, a_l, a_r, out_acc);
    else
      edge_scatter<32, 4><<<E, 128, 0, stream>>>(src, dst, hidden, norm, a_l, a_r, out_acc);

    // snorm scale + BN (train mode, batch stats) + LeakyReLU(0.2) -> f16 feats
    bn_stats<<<128, 256, 0, stream>>>(out_acc, snorm, mean, var, N);
    bn_apply<<<(tot + 255) / 256, 256, 0, stream>>>(out_acc, snorm, mean, var,
                                                    bng[layer], bnb[layer], feat_h, tot);
  }

  // Classifier head: h1 = relu(relu(feat) @ cls1_W + b1)   (reuse out_acc as h1)
  stage_b_kernel<<<(128 * 64 + 255) / 256, 256, 0, stream>>>(cls1W, wh, 128, 64);
  float* h1 = out_acc;
  dim3 g1(gemm_gx, 64 / 16);
  gemm_wmma<true, true><<<g1, 128, 0, stream>>>(feat_h, wh, cls1b, h1, N, 128, 64);

  // logits = h1 @ cls2_W + b2
  cls2_kernel<<<(N * 2 + 255) / 256, 256, 0, stream>>>(h1, cls2W, cls2b, (float*)d_out, N * 2);
}
